// QuantumLinear_56624848831099
// MI455X (gfx1250) — matrix operand order repaired and checked by reading, not compiled, not run on hardware
//
#include <hip/hip_runtime.h>

// out = x @ Wr^T + bias     (exact identity: magnitude*cos(atan2(i,r)) == r)
// Single f16 WMMA GEMM, f32 accumulate, software-pipelined ping-pong LDS.

typedef _Float16 v16h __attribute__((ext_vector_type(16)));
typedef _Float16 v8h  __attribute__((ext_vector_type(8)));
typedef _Float16 v4h  __attribute__((ext_vector_type(4)));
typedef float    v8f  __attribute__((ext_vector_type(8)));
typedef float    v4f  __attribute__((ext_vector_type(4)));

constexpr int IN_F  = 4096;
constexpr int OUT_F = 4096;
constexpr int ROWS  = 8192;

constexpr int BM = 128;          // block tile M
constexpr int BN = 128;          // block tile N
constexpr int BK = 32;           // K step (one 16x16x32 WMMA deep)
constexpr int KPAD = 8;          // LDS pad: row stride 40 halfs = 80 B (20 banks)
constexpr int LDSS = BK + KPAD;

__global__ __launch_bounds__(256)
void qlinear_wmma_kernel(const float* __restrict__ x,
                         const float* __restrict__ wr,
                         const float* __restrict__ bias,
                         float* __restrict__ out)
{
    __shared__ _Float16 As[2][BM * LDSS];   // ping-pong x tiles, f16, [row][k]
    __shared__ _Float16 Bs[2][BN * LDSS];   // ping-pong Wr tiles, f16, [n][k]

    const int tid      = threadIdx.x;
    const int lane     = tid & 31;
    const int wave     = tid >> 5;
    const int halfLane = lane & 15;
    const int hiHalf   = lane >> 4;      // 0: lanes 0-15, 1: lanes 16-31

    const int wave_m = wave & 3;         // 4 waves along M, 32 rows each
    const int wave_n = wave >> 2;        // 2 waves along N, 64 cols each

    const int m0 = blockIdx.y * BM;
    const int n0 = blockIdx.x * BN;

    v8f acc[2][4] = {};                  // 2 M-frags x 4 N-frags of 16x16 f32

    v4f areg[4], breg[4];                // in-flight staging registers

    auto load_global = [&](int k0) {
        #pragma unroll
        for (int i = 0; i < 4; ++i) {
            const int idx = i * 256 + tid;       // 0..1023
            const int row = idx >> 3;            // 0..127
            const int kq  = (idx & 7) << 2;      // 0,4,...,28
            const size_t ga = (size_t)(m0 + row) * IN_F + k0 + kq;
            const size_t gb = (size_t)(n0 + row) * IN_F + k0 + kq;
            areg[i] = *reinterpret_cast<const v4f*>(&x[ga]);
            breg[i] = *reinterpret_cast<const v4f*>(&wr[gb]);
            if (i == 0 && k0 + BK < IN_F) {      // L2 prefetch two tiles ahead of compute
                __builtin_prefetch(&x[ga + BK], 0, 1);
                __builtin_prefetch(&wr[gb + BK], 0, 1);
            }
        }
    };

    auto store_lds = [&](int buf) {
        #pragma unroll
        for (int i = 0; i < 4; ++i) {
            const int idx = i * 256 + tid;
            const int row = idx >> 3;
            const int kq  = (idx & 7) << 2;
            v4h ah = { (_Float16)areg[i][0], (_Float16)areg[i][1],
                       (_Float16)areg[i][2], (_Float16)areg[i][3] };
            v4h bh = { (_Float16)breg[i][0], (_Float16)breg[i][1],
                       (_Float16)breg[i][2], (_Float16)breg[i][3] };
            *reinterpret_cast<v4h*>(&As[buf][row * LDSS + kq]) = ah;   // 8B aligned
            *reinterpret_cast<v4h*>(&Bs[buf][row * LDSS + kq]) = bh;
        }
    };

    auto compute = [&](int buf) {
        // A frag 16x32 f16: lane<16 holds K0-7,16-23; lane>=16 holds K8-15,24-31
        v16h afrag[2];
        #pragma unroll
        for (int mi = 0; mi < 2; ++mi) {
            const _Float16* base = &As[buf][(wave_m * 32 + mi * 16 + halfLane) * LDSS];
            v8h lo = *reinterpret_cast<const v8h*>(base + hiHalf * 8);       // 16B aligned
            v8h hi = *reinterpret_cast<const v8h*>(base + 16 + hiHalf * 8);
            afrag[mi] = __builtin_shufflevector(lo, hi,
                0,1,2,3,4,5,6,7,8,9,10,11,12,13,14,15);
        }
        // B frag 32x16: lane n holds col n of B = row n of Wr, same K split as A
        v16h bfrag[4];
        #pragma unroll
        for (int ni = 0; ni < 4; ++ni) {
            const _Float16* base =
                &Bs[buf][(wave_n * 64 + ni * 16 + halfLane) * LDSS + hiHalf * 8];
            v8h lo = *reinterpret_cast<const v8h*>(base);                    // 16B aligned
            v8h hi = *reinterpret_cast<const v8h*>(base + 16);
            bfrag[ni] = __builtin_shufflevector(lo, hi,
                0,1,2,3,4,5,6,7,8,9,10,11,12,13,14,15);
        }
        #pragma unroll
        for (int mi = 0; mi < 2; ++mi) {
            #pragma unroll
            for (int ni = 0; ni < 4; ++ni) {
                acc[mi][ni] = __builtin_amdgcn_wmma_f32_16x16x32_f16(
                    false, afrag[mi], false, bfrag[ni],
                    (short)0, acc[mi][ni], false, false);
            }
        }
    };

    // ---- pipelined main loop: 1 barrier per K-step ----
    load_global(0);
    store_lds(0);

    #pragma unroll 2
    for (int k0 = 0; k0 < IN_F; k0 += BK) {
        const int buf = (k0 >> 5) & 1;           // const-folds under unroll-2
        __syncthreads();                         // LDS[buf] ready; prev readers of buf^1 done
        if (k0 + BK < IN_F) load_global(k0 + BK);
        compute(buf);
        if (k0 + BK < IN_F) store_lds(buf ^ 1);
    }

    // ---- epilogue: C/D layout. VGPR v, lane<16: M=v, N=lane; lane>=16: M=8+v ----
    #pragma unroll
    for (int ni = 0; ni < 4; ++ni) {
        const int gn = n0 + wave_n * 64 + ni * 16 + halfLane;
        const float bv = bias[gn];
        #pragma unroll
        for (int mi = 0; mi < 2; ++mi) {
            const int gm = m0 + wave_m * 32 + mi * 16 + hiHalf * 8;
            #pragma unroll
            for (int v = 0; v < 8; ++v) {
                out[(size_t)(gm + v) * OUT_F + gn] = acc[mi][ni][v] + bv;
            }
        }
    }
}

extern "C" void kernel_launch(void* const* d_in, const int* in_sizes, int n_in,
                              void* d_out, int out_size, void* d_ws, size_t ws_size,
                              hipStream_t stream) {
    const float* x    = (const float*)d_in[0];   // [ROWS, IN_F]
    const float* wr   = (const float*)d_in[1];   // [OUT_F, IN_F]
    // d_in[2] = weight_imag: mathematically cancels (mag*cos(phase) == real)
    const float* bias = (const float*)d_in[3];   // [OUT_F]
    float* out = (float*)d_out;                  // [ROWS, OUT_F]

    dim3 grid(OUT_F / BN, ROWS / BM);            // 32 x 64 blocks
    dim3 block(256);                             // 8 waves (wave32)
    hipLaunchKernelGGL(qlinear_wmma_kernel, grid, block, 0, stream, x, wr, bias, out);
}
